// RNN_20289425506787
// MI455X (gfx1250) — matrix operand order repaired and checked by reading, not compiled, not run on hardware
//
#include <hip/hip_runtime.h>
#include <math.h>

typedef __attribute__((ext_vector_type(16))) _Float16     v16h;
typedef __attribute__((ext_vector_type(8)))  float        v8f;
typedef __attribute__((ext_vector_type(4)))  unsigned int u32x4;
typedef __attribute__((ext_vector_type(8)))  int          i32x8;
typedef __attribute__((ext_vector_type(4)))  int          i32x4;

#define B_TOT    4096
#define T_TOT    512
#define DH       128
#define M_TILE   32
#define TCH      64
#define NTHREADS 256
#define HSZ      (M_TILE * DH)

#if defined(__gfx1250__) && __has_builtin(__builtin_amdgcn_tensor_load_to_lds)
#define USE_TDM 1
#else
#define USE_TDM 0
#endif

#if __has_include(<hip/amd_detail/amd_gfx1250_TDM.h>)
#define TDM_SIX_ARG 1
#endif

#define WMMA_F16(A, Bf, C) \
    __builtin_amdgcn_wmma_f32_16x16x32_f16(false, (A), false, (Bf), (short)0, (C), false, false)

__device__ __forceinline__ float fast_tanh(float x) {
#if defined(__gfx1250__)
  #if __has_builtin(__builtin_amdgcn_tanhf)
    return __builtin_amdgcn_tanhf(x);
  #else
    const float e = __builtin_amdgcn_exp2f(x * 2.8853900817779268f);
    return 1.0f - 2.0f * __builtin_amdgcn_rcpf(e + 1.0f);
  #endif
#else
    return tanhf(x);
#endif
}

// Hidden-dim storage permutation: storage column (pair*32 + 2q + b) holds
// hidden unit (pair*32 + b*16 + q), i.e. within each 32-wide pair of column
// blocks the two blocks are interleaved. This makes each lane's two output
// units adjacent halves -> one packed b32 store.
__device__ __forceinline__ int perm_unit(int kstore) {
    const int pair   = kstore >> 5;
    const int within = kstore & 31;
    return pair * 32 + (within & 1) * 16 + (within >> 1);
}

__global__ __launch_bounds__(NTHREADS) void rnn_elman_wmma(
    const float* __restrict__ seq,    // (B, T)
    const float* __restrict__ Wh_w,   // (DH, DH)
    const float* __restrict__ Wh_b,   // (DH)
    const float* __restrict__ Wi_w,   // (DH, 1)
    const float* __restrict__ Wi_b,   // (DH)
    const float* __restrict__ Wo_w,   // (DH, DH)
    const float* __restrict__ Wo_b,   // (DH)
    float* __restrict__ out)          // (B, DH) softmax
{
    __shared__ _Float16 h_sh[2 * HSZ];           // 16 KB : ping-pong recurrent state
    __shared__ float    x_sh[TCH * M_TILE];      //  8 KB : seq chunk, [t][row]
    __shared__ float    logit_sh[M_TILE * DH];   // 16 KB : TDM staging, then logits

    const int tid     = threadIdx.x;
    const int wave    = tid >> 5;
    const int lane    = tid & 31;
    const int rowbase = blockIdx.x * M_TILE;

    const int r    = wave & 1;            // row-block (0..1) of this wave
    const int cb   = (wave >> 1) << 1;    // first of 2 col-blocks (0,2,4,6)
    const int nloc = lane & 15;
    const int hi   = lane >> 4;

    // ---- Wh^T (B-operand) fragments resident in VGPRs; K enumerates the
    // PERMUTED hidden storage order so A-loads from h stay contiguous.
    v16h bfrag[2][4];
    float biasn[2], win[2], wob[2];
    #pragma unroll
    for (int j = 0; j < 2; ++j) {
        const int n = (cb + j) * 16 + nloc;   // true output unit of this column
        #pragma unroll
        for (int kc = 0; kc < 4; ++kc) {
            v16h f;
            #pragma unroll
            for (int v = 0; v < 16; ++v)
                f[v] = (_Float16)Wh_w[n * DH + perm_unit(kc * 32 + hi * 16 + v)];
            bfrag[j][kc] = f;
        }
        biasn[j] = Wh_b[n] + Wi_b[n];
        win[j]   = Wi_w[n];          // D_IN == 1
        wob[j]   = Wo_b[n];
    }

    for (int i = tid; i < HSZ; i += NTHREADS) h_sh[i] = (_Float16)0.f;
    __syncthreads();

    const int moff  = r * 16 + hi * 8;        // D rows this lane owns: moff + v
    const int am    = r * 16 + nloc;          // A-matrix row for this lane
    const int akoff = hi * 8;                 // unused; A loads offset K by hi*16 per half-wave
    const int scol  = cb * 16 + 2 * nloc;     // packed storage column (even)

    _Float16* const hA = h_sh;
    _Float16* const hB = h_sh + HSZ;

    // One recurrence step: h_dst = tanh(h_src @ Wh^T + bias + x*wi)
    auto rnn_step = [&](const _Float16* hsrc, _Float16* hdst, int tl) {
        // x for this lane's 8 rows: two broadcast b128 LDS loads
        const float* xp = x_sh + tl * M_TILE + moff;
        const float4 xa = *(const float4*)(xp);
        const float4 xb = *(const float4*)(xp + 4);
        const float xv[8] = {xa.x, xa.y, xa.z, xa.w, xb.x, xb.y, xb.z, xb.w};
        // bias + x*wi folded into the WMMA C operand
        v8f acc0, acc1;
        #pragma unroll
        for (int v = 0; v < 8; ++v) {
            acc0[v] = fmaf(xv[v], win[0], biasn[0]);
            acc1[v] = fmaf(xv[v], win[1], biasn[1]);
        }
        #pragma unroll
        for (int kc = 0; kc < 4; ++kc) {
            // A layout (16x32 f16): lanes 0-15 K {0..15}; 16-31 K {16..31}
            v16h a;
            #pragma unroll
            for (int v = 0; v < 8; ++v) {
                const int kk = kc * 32 + hi * 16 + 2 * v;
                a[2 * v]     = hsrc[am * DH + kk];
                a[2 * v + 1] = hsrc[am * DH + kk + 1];
            }
            acc0 = WMMA_F16(a, bfrag[0][kc], acc0);   // D->C chaining: no hazard NOPs
            acc1 = WMMA_F16(a, bfrag[1][kc], acc1);
        }
        #pragma unroll
        for (int v = 0; v < 8; ++v) {
            const int m = moff + v;
            union { _Float16 h2[2]; unsigned u; } pk;
            pk.h2[0] = (_Float16)fast_tanh(acc0[v]);
            pk.h2[1] = (_Float16)fast_tanh(acc1[v]);
            *(unsigned*)((void*)&hdst[m * DH + scol]) = pk.u;   // one b32 store
        }
        __syncthreads();
    };

    for (int t0 = 0; t0 < T_TOT; t0 += TCH) {
#if USE_TDM
        // ---- TDM: DMA 32 rows x 64 timesteps of seq into LDS staging
        if (wave == 0) {
            const unsigned lds_off =
                (unsigned)(unsigned long long)(const void*)logit_sh;
            const unsigned long long ga =
                (unsigned long long)(const void*)(seq + (size_t)rowbase * T_TOT + t0);
            u32x4 g0 = {};
            g0[0] = 1u;                                   // count=1 (valid user D#)
            g0[1] = lds_off;                              // lds_addr
            g0[2] = (unsigned)(ga & 0xFFFFFFFFull);       // global_addr[31:0]
            g0[3] = (unsigned)((ga >> 32) & 0x1FFFFFFull) // global_addr[56:32]
                    | (2u << 30);                         // type=2 (image)
            i32x8 g1 = {};
            g1[0] = 2 << 16;                              // data_size=4B, wg_mask=0
            g1[1] = TCH << 16;                            // tensor_dim0 = 64
            g1[2] = M_TILE << 16;                         // tensor_dim1 = 32
            g1[3] = TCH << 16;                            // tile_dim0 = 64
            g1[4] = M_TILE;                               // tile_dim1 = 32, tile_dim2=0
            g1[5] = T_TOT;                                // tensor_dim0_stride = 512
            i32x4 gz = {};
#if defined(TDM_SIX_ARG)
            i32x8 gz8 = {};
            __builtin_amdgcn_tensor_load_to_lds(g0, g1, gz, gz, gz8, 0);
#else
            __builtin_amdgcn_tensor_load_to_lds(g0, g1, gz, gz, 0);
#endif
            __builtin_amdgcn_s_wait_tensorcnt(0);
        }
        __syncthreads();
        for (int i = tid; i < M_TILE * TCH; i += NTHREADS) {
            const int row = i / TCH;
            const int t   = i % TCH;
            x_sh[t * M_TILE + row] = logit_sh[i];   // [row][t] -> [t][row]
        }
        __syncthreads();
#else
        // ---- fallback: coalesced float4 loads, store transposed
        #pragma unroll
        for (int q = 0; q < 2; ++q) {
            const int f4  = tid * 2 + q;
            const int row = f4 >> 4;
            const int c4  = f4 & 15;
            const float4 v4 =
                *(const float4*)(seq + (size_t)(rowbase + row) * T_TOT + t0 + c4 * 4);
            x_sh[(c4 * 4 + 0) * M_TILE + row] = v4.x;
            x_sh[(c4 * 4 + 1) * M_TILE + row] = v4.y;
            x_sh[(c4 * 4 + 2) * M_TILE + row] = v4.z;
            x_sh[(c4 * 4 + 3) * M_TILE + row] = v4.w;
        }
        if (t0 + TCH < T_TOT)
            __builtin_prefetch(seq + (size_t)(rowbase + (tid >> 3)) * T_TOT
                               + (t0 + TCH) + (tid & 7) * 8, 0, 1);
        __syncthreads();
#endif
        // ---- 64 recurrence steps, ping-pong h buffers, 1 barrier/step
        for (int tl = 0; tl < TCH; tl += 2) {
            rnn_step(hA, hB, tl);
            rnn_step(hB, hA, tl + 1);
        }
    }

    // ---- logits = h_T @ Wo^T + Wo_b  (h_T in hA; Wo fragments use same perm)
    {
        v8f acc0, acc1;
        #pragma unroll
        for (int v = 0; v < 8; ++v) { acc0[v] = wob[0]; acc1[v] = wob[1]; }
        #pragma unroll
        for (int kc = 0; kc < 4; ++kc) {
            v16h a;
            #pragma unroll
            for (int v = 0; v < 8; ++v) {
                const int kk = kc * 32 + hi * 16 + 2 * v;
                a[2 * v]     = hA[am * DH + kk];
                a[2 * v + 1] = hA[am * DH + kk + 1];
            }
            #pragma unroll
            for (int j = 0; j < 2; ++j) {
                const int n = (cb + j) * 16 + nloc;
                v16h f;
                #pragma unroll
                for (int v = 0; v < 16; ++v)
                    f[v] = (_Float16)Wo_w[n * DH + perm_unit(kc * 32 + hi * 16 + v)];
                if (j == 0) acc0 = WMMA_F16(a, f, acc0);
                else        acc1 = WMMA_F16(a, f, acc1);
            }
        }
        __syncthreads();   // staging uses of logit_sh are done
        #pragma unroll
        for (int v = 0; v < 8; ++v) {
            const int m = moff + v;
            logit_sh[m * DH + (cb + 0) * 16 + nloc] = acc0[v];
            logit_sh[m * DH + (cb + 1) * 16 + nloc] = acc1[v];
        }
    }
    __syncthreads();

    // ---- row softmax (tail; one thread per batch row)
    if (tid < M_TILE) {
        const int m = tid;
        float mx = -INFINITY;
        #pragma unroll 4
        for (int n = 0; n < DH; ++n) mx = fmaxf(mx, logit_sh[m * DH + n]);
        float s = 0.f;
        #pragma unroll 4
        for (int n = 0; n < DH; ++n) {
            const float e = expf(logit_sh[m * DH + n] - mx);
            logit_sh[m * DH + n] = e;
            s += e;
        }
        const float inv = 1.f / s;
        float* orow = out + (size_t)(rowbase + m) * DH;
        #pragma unroll 4
        for (int n = 0; n < DH; ++n) orow[n] = logit_sh[m * DH + n] * inv;
    }
}

extern "C" void kernel_launch(void* const* d_in, const int* in_sizes, int n_in,
                              void* d_out, int out_size, void* d_ws, size_t ws_size,
                              hipStream_t stream) {
    (void)in_sizes; (void)n_in; (void)out_size; (void)d_ws; (void)ws_size;
    const float* seq  = (const float*)d_in[0];
    const float* Wh_w = (const float*)d_in[1];
    const float* Wh_b = (const float*)d_in[2];
    const float* Wi_w = (const float*)d_in[3];
    const float* Wi_b = (const float*)d_in[4];
    const float* Wo_w = (const float*)d_in[5];
    const float* Wo_b = (const float*)d_in[6];
    float* out = (float*)d_out;

    dim3 grid(B_TOT / M_TILE);   // 128 workgroups
    dim3 block(NTHREADS);        // 8 wave32 waves
    rnn_elman_wmma<<<grid, block, 0, stream>>>(seq, Wh_w, Wh_b, Wi_w, Wi_b, Wo_w, Wo_b, out);
}
